// HeteNet_58969900974561
// MI455X (gfx1250) — matrix operand order repaired and checked by reading, not compiled, not run on hardware
//
#include <hip/hip_runtime.h>

typedef __attribute__((ext_vector_type(16))) _Float16 v16h;
typedef __attribute__((ext_vector_type(8)))  _Float16 v8h;
typedef __attribute__((ext_vector_type(8)))  float    v8f;

#define NTP 4
#define NNET 5      // 4 experts + central critic
#define AG  128     // agents per timestep
#define D   64
#define H   64
#define NA  16
#define OUTS (NA + 1)   // 17

#define WMMA_F16(a, b, c) \
  __builtin_amdgcn_wmma_f32_16x16x32_f16(false, (a), false, (b), (short)0, (c), false, false)

// A fragment (16x32 f16, row-major tile with row stride 64 halves):
// lane l<16 : row l, K = k0..k0+15
// lane l>=16: row l-16, K = k0+16..k0+31
__device__ __forceinline__ v16h load_a_frag(const _Float16* row, int hi, int k0) {
  v8h lo = *(const v8h*)(row + k0 + hi * 16);
  v8h hh = *(const v8h*)(row + k0 + 8 + hi * 16);
  return __builtin_shufflevector(lo, hh, 0,1,2,3,4,5,6,7,8,9,10,11,12,13,14,15);
}

__global__ __launch_bounds__(256) void hetenet_wmma_kernel(
    const float* __restrict__ obs,   // [T,128,64]
    const int*   __restrict__ pick,  // [T,128]
    const float* __restrict__ gp,    // [T,4]
    const float* __restrict__ W1,    // [4,68,64]
    const float* __restrict__ b1,    // [4,64]
    const float* __restrict__ W2,    // [4,64,64]
    const float* __restrict__ b2,    // [4,64]
    const float* __restrict__ W3,    // [4,64,16]
    const float* __restrict__ b3,    // [4,16]
    const float* __restrict__ cW1,   // [68,64]
    const float* __restrict__ cb1,   // [64]
    const float* __restrict__ cW2,   // [64,64]
    const float* __restrict__ cb2,   // [64]
    const float* __restrict__ cW3,   // [64,1]
    const float* __restrict__ cb3,   // [1]
    float*       __restrict__ out)   // [T,128,17]
{
  const int t    = blockIdx.x;       // one timestep per block
  const int tid  = threadIdx.x;      // 0..255, 8 waves
  const int wave = tid >> 5;
  const int lane = tid & 31;
  const int r    = lane & 15;        // column / row-within-half
  const int hi   = lane >> 4;        // half select

  __shared__ __align__(32) _Float16 sX[AG][D];       // 16 KB  obs tile (f16)
  __shared__ __align__(32) _Float16 sW1t[H][D];      //  8 KB  W1^T[h][k], current net
  __shared__ __align__(32) _Float16 sW2t[H][H];      //  8 KB  W2^T
  __shared__ __align__(32) _Float16 sW3t[NA][H];     //  2 KB  W3^T (critic padded to 16)
  __shared__ __align__(32) _Float16 sHt[8][16][H];   // 16 KB  per-wave hidden tiles
  __shared__ float sB1[H], sB2[H], sB3[NA];
  __shared__ float sGP[NTP];

  if (tid < NTP) sGP[tid] = gp[t * NTP + tid];
  {
    const float* src = obs + (size_t)t * AG * D;
    _Float16* dst = &sX[0][0];
    for (int i = tid; i < AG * D; i += 256) dst[i] = (_Float16)src[i];
  }

  const int tokBase = t * AG + wave * 16;
  // per-lane expert picks for the 8 output rows this lane owns (m = hi*8 + j)
  int myPick[8];
  #pragma unroll
  for (int j = 0; j < 8; ++j) myPick[j] = pick[tokBase + hi * 8 + j];

  __syncthreads();

  for (int net = 0; net < NNET; ++net) {
    __syncthreads();  // previous net's compute fully done before re-staging
    // ---------------- stage weights (f16, transposed) + folded biases ----------------
    if (net < NTP) {
      const float* w1 = W1 + (size_t)net * (D + NTP) * H;  // [68][64]
      const float* w2 = W2 + (size_t)net * H * H;          // [64][64]
      const float* w3 = W3 + (size_t)net * H * NA;         // [64][16]
      for (int i = tid; i < H * D; i += 256) {
        int h = i >> 6, k = i & 63;
        sW1t[h][k] = (_Float16)w1[k * H + h];
        sW2t[h][k] = (_Float16)w2[k * H + h];
      }
      for (int i = tid; i < NA * H; i += 256) {
        int n = i >> 6, k = i & 63;
        sW3t[n][k] = (_Float16)w3[k * NA + n];
      }
      if (tid < H) {
        float acc = b1[net * H + tid];
        #pragma unroll
        for (int i = 0; i < NTP; ++i) acc += sGP[i] * w1[(D + i) * H + tid];
        sB1[tid] = acc;
        sB2[tid] = b2[net * H + tid];
      }
      if (tid < NA) sB3[tid] = b3[net * NA + tid];
    } else {  // central critic, output padded 1 -> 16 (only column 0 stored)
      for (int i = tid; i < H * D; i += 256) {
        int h = i >> 6, k = i & 63;
        sW1t[h][k] = (_Float16)cW1[k * H + h];
        sW2t[h][k] = (_Float16)cW2[k * H + h];
      }
      for (int i = tid; i < NA * H; i += 256) {
        int n = i >> 6, k = i & 63;
        sW3t[n][k] = (n == 0) ? (_Float16)cW3[k] : (_Float16)0.0f;
      }
      if (tid < H) {
        float acc = cb1[tid];
        #pragma unroll
        for (int i = 0; i < NTP; ++i) acc += sGP[i] * cW1[(D + i) * H + tid];
        sB1[tid] = acc;
        sB2[tid] = cb2[tid];
      }
      if (tid < NA) sB3[tid] = (tid == 0) ? cb3[0] : 0.0f;
    }
    __syncthreads();

    const _Float16* xt = &sX[wave * 16][0];     // [16][64] A tile
    _Float16*       ht = &sHt[wave][0][0];      // [16][64] hidden tile

    // ---------------- layer 1: x(16x64) @ W1(64x64) -> h1 ----------------
    v8f acc1[4] = {};
    #pragma unroll
    for (int k0 = 0; k0 < D; k0 += 32) {
      v16h a = load_a_frag(xt + r * D, hi, k0);
      #pragma unroll
      for (int nt = 0; nt < 4; ++nt) {
        v16h b = *(const v16h*)(&sW1t[nt * 16 + r][0] + k0 + hi * 16);
        acc1[nt] = WMMA_F16(a, b, acc1[nt]);
      }
    }
    #pragma unroll
    for (int nt = 0; nt < 4; ++nt) {
      const float bias = sB1[nt * 16 + r];
      #pragma unroll
      for (int j = 0; j < 8; ++j) {
        float v = acc1[nt][j] + bias;
        ht[(hi * 8 + j) * H + nt * 16 + r] = (_Float16)(v > 0.0f ? v : 0.0f);
      }
    }

    // ---------------- layer 2: h1(16x64) @ W2(64x64) -> h2 ----------------
    v8f acc2[4] = {};
    #pragma unroll
    for (int k0 = 0; k0 < H; k0 += 32) {
      v16h a = load_a_frag(ht + r * H, hi, k0);
      #pragma unroll
      for (int nt = 0; nt < 4; ++nt) {
        v16h b = *(const v16h*)(&sW2t[nt * 16 + r][0] + k0 + hi * 16);
        acc2[nt] = WMMA_F16(a, b, acc2[nt]);
      }
    }
    #pragma unroll
    for (int nt = 0; nt < 4; ++nt) {
      const float bias = sB2[nt * 16 + r];
      #pragma unroll
      for (int j = 0; j < 8; ++j) {
        float v = acc2[nt][j] + bias;
        ht[(hi * 8 + j) * H + nt * 16 + r] = (_Float16)(v > 0.0f ? v : 0.0f);
      }
    }

    // ---------------- layer 3: h2(16x64) @ W3(64x16) -> head ----------------
    v8f acc3 = {};
    #pragma unroll
    for (int k0 = 0; k0 < H; k0 += 32) {
      v16h a = load_a_frag(ht + r * H, hi, k0);
      v16h b = *(const v16h*)(&sW3t[r][0] + k0 + hi * 16);
      acc3 = WMMA_F16(a, b, acc3);
    }

    // ---------------- scatter outputs ----------------
    if (net < NTP) {
      const float bias = sB3[r];
      #pragma unroll
      for (int j = 0; j < 8; ++j) {
        if (myPick[j] == net) {
          int token = tokBase + hi * 8 + j;
          out[(size_t)token * OUTS + r] = acc3[j] + bias;
        }
      }
    } else if (r == 0) {  // critic value -> slot 16 (lanes 0 and 16 cover m=0..15)
      const float bias = sB3[0];
      #pragma unroll
      for (int j = 0; j < 8; ++j) {
        int token = tokBase + hi * 8 + j;
        out[(size_t)token * OUTS + NA] = acc3[j] + bias;
      }
    }
  }
}

extern "C" void kernel_launch(void* const* d_in, const int* in_sizes, int n_in,
                              void* d_out, int out_size, void* d_ws, size_t ws_size,
                              hipStream_t stream) {
  const float* obs  = (const float*)d_in[0];
  const int*   pick = (const int*)  d_in[1];
  const float* gp   = (const float*)d_in[2];
  const float* W1   = (const float*)d_in[3];
  const float* b1   = (const float*)d_in[4];
  const float* W2   = (const float*)d_in[5];
  const float* b2   = (const float*)d_in[6];
  const float* W3   = (const float*)d_in[7];
  const float* b3   = (const float*)d_in[8];
  const float* cW1  = (const float*)d_in[9];
  const float* cb1  = (const float*)d_in[10];
  const float* cW2  = (const float*)d_in[11];
  const float* cb2  = (const float*)d_in[12];
  const float* cW3  = (const float*)d_in[13];
  const float* cb3  = (const float*)d_in[14];
  float* out = (float*)d_out;

  const int T = in_sizes[0] / (AG * D);   // 2048
  hipLaunchKernelGGL(hetenet_wmma_kernel, dim3(T), dim3(256), 0, stream,
                     obs, pick, gp, W1, b1, W2, b2, W3, b3,
                     cW1, cb1, cW2, cb2, cW3, cb3, out);
}
